// DigitConvolutionalModel_71571335020722
// MI455X (gfx1250) — matrix operand order repaired and checked by reading, not compiled, not run on hardware
//
#include <hip/hip_runtime.h>
#include <hip/hip_bf16.h>

// ---------------------------------------------------------------------------
// Fused MNIST-ish model for MI455X (gfx1250, wave32):
//   conv3x3(valid) -> [B,676] -> @w1[676,300]+b1, relu -> @w2[300,10]+b2
// Strategy: bf16 WMMA (f32 accumulate) for both GEMMs, all intermediates in
// LDS (never spilled to HBM), weights pre-transposed/padded to bf16 in d_ws.
// ---------------------------------------------------------------------------

typedef __attribute__((ext_vector_type(16))) __bf16 v16bf;
typedef __attribute__((ext_vector_type(8)))  float  v8f;

#define TILE_M 128      // rows per workgroup (8 waves x 16-row M tiles)
#define K1P    704      // 676 padded to 22*32
#define N1P    304      // 300 padded to 19*16
#define K2P    320      // 300 padded to 10*32

__device__ __forceinline__ unsigned short f2bf(float f) {
    unsigned int u = __float_as_uint(f);
    u = (u + 0x7FFFu + ((u >> 16) & 1u)) >> 16;   // round-to-nearest-even
    return (unsigned short)u;
}

union Frag16 { v16bf v; uint4 q[2]; };

// Build an A fragment (16-bit 16x32 layout): per lane, two contiguous 8-elem
// chunks: K = k0 + hi*8 .. +7  and  K = k0 + 16 + hi*8 .. +7   (hi = lane>=16)
__device__ __forceinline__ v16bf load_a_frag(const unsigned short* rowp, int k0, int hi) {
    Frag16 u;
    u.q[0] = *(const uint4*)(rowp + k0 + hi * 8);
    u.q[1] = *(const uint4*)(rowp + k0 + 16 + hi * 8);
    return u.v;
}

// Build a B fragment (16-bit 32x16 layout): per lane (column n = lane&15),
// two 8-elem chunks: K = k0 + hi*8 .. +7  and  K = k0 + 16 + hi*8 .. +7.
__device__ __forceinline__ v16bf load_b_frag(const unsigned short* colp, int k0, int hi) {
    Frag16 u;
    u.q[0] = *(const uint4*)(colp + k0 + hi * 8);
    u.q[1] = *(const uint4*)(colp + k0 + 16 + hi * 8);
    return u.v;
}

// --- weight prep: w1[676,300] f32 -> w1T[304][704] bf16 (zero padded) ------
__global__ void prep_w1_kernel(const float* __restrict__ w1, unsigned short* __restrict__ w1T) {
    int i = blockIdx.x * 256 + threadIdx.x;
    if (i >= N1P * K1P) return;
    int n = i / K1P, k = i - n * K1P;
    float v = (n < 300 && k < 676) ? w1[k * 300 + n] : 0.0f;
    w1T[i] = f2bf(v);
}

// --- weight prep: w2[300,10] f32 -> w2T[16][320] bf16 (zero padded) --------
__global__ void prep_w2_kernel(const float* __restrict__ w2, unsigned short* __restrict__ w2T) {
    int i = blockIdx.x * 256 + threadIdx.x;
    if (i >= 16 * K2P) return;
    int n = i / K2P, k = i - n * K2P;
    float v = (n < 10 && k < 300) ? w2[k * 10 + n] : 0.0f;
    w2T[i] = f2bf(v);
}

// --- fused conv + GEMM1(relu) + GEMM2 --------------------------------------
__global__ __launch_bounds__(256)
void fused_digit_model_kernel(const float* __restrict__ x,
                              const float* __restrict__ conv_w,
                              const float* __restrict__ b1,
                              const float* __restrict__ b2,
                              const unsigned short* __restrict__ w1T,
                              const unsigned short* __restrict__ w2T,
                              float* __restrict__ out,
                              int B) {
    extern __shared__ unsigned short smem[];
    unsigned short* s_feat = smem;                    // [128][704] bf16 (180224 B)
    unsigned short* s_a1   = smem + TILE_M * K1P;     // [128][320] bf16 ( 81920 B)

    const int t = threadIdx.x;
    const long long rowBase = (long long)blockIdx.x * TILE_M;

    float cw[9];
#pragma unroll
    for (int i = 0; i < 9; ++i) cw[i] = conv_w[i];

    // ---- Phase 1: 3x3 valid conv -> bf16 feat in LDS; zero a1 pad region --
    for (int i = t; i < TILE_M * K1P; i += 256) {
        int r = i / K1P, c = i - r * K1P;
        float v = 0.0f;
        long long gr = rowBase + r;
        if (c < 676 && gr < B) {
            int oy = c / 26, ox = c - oy * 26;
            const float* xp = x + gr * 784 + oy * 28 + ox;
#pragma unroll
            for (int ky = 0; ky < 3; ++ky)
#pragma unroll
                for (int kx = 0; kx < 3; ++kx)
                    v = fmaf(xp[ky * 28 + kx], cw[ky * 3 + kx], v);
        }
        s_feat[i] = f2bf(v);
    }
    for (int i = t; i < TILE_M * K2P; i += 256) s_a1[i] = 0;
    __syncthreads();

    const int wave = t >> 5;            // 0..7 -> owns rows [wave*16, wave*16+16)
    const int lane = t & 31;
    const int ln   = lane & 15;         // column within 16-wide N tile
    const int hi   = lane >> 4;         // half-wave selector
    const int m0   = wave * 16;

    // ---- Phase 2: feat[128,704] @ w1T -> relu -> a1 bf16 in LDS -----------
    const unsigned short* ap1 = s_feat + (m0 + ln) * K1P;
    for (int j = 0; j < N1P / 16; ++j) {
        const int n = j * 16 + ln;
        const unsigned short* bp = w1T + n * K1P;
        v8f acc = {};
#pragma unroll 2
        for (int k0 = 0; k0 < K1P; k0 += 32) {
            v16bf a = load_a_frag(ap1, k0, hi);
            v16bf b = load_b_frag(bp, k0, hi);
            acc = __builtin_amdgcn_wmma_f32_16x16x32_bf16(
                false, a, false, b, (short)0, acc, false, false);
        }
        const float bias = (n < 300) ? b1[n] : 0.0f;
#pragma unroll
        for (int v = 0; v < 8; ++v) {
            int m = m0 + (hi ? 8 + v : v);          // C/D layout: vgpr v -> row
            float r = acc[v] + bias;
            r = r > 0.0f ? r : 0.0f;                // relu
            s_a1[m * K2P + n] = f2bf(r);
        }
    }
    __syncthreads();

    // ---- Phase 3: a1[128,320] @ w2T[16][320] -> out[B,10] -----------------
    {
        const unsigned short* ap2 = s_a1 + (m0 + ln) * K2P;
        const unsigned short* bp  = w2T + ln * K2P;
        v8f acc = {};
#pragma unroll
        for (int k0 = 0; k0 < K2P; k0 += 32) {
            v16bf a = load_a_frag(ap2, k0, hi);
            v16bf b = load_b_frag(bp, k0, hi);
            acc = __builtin_amdgcn_wmma_f32_16x16x32_bf16(
                false, a, false, b, (short)0, acc, false, false);
        }
        if (ln < 10) {
            const float bias = b2[ln];
#pragma unroll
            for (int v = 0; v < 8; ++v) {
                int m = m0 + (hi ? 8 + v : v);
                long long gr = rowBase + m;
                if (gr < B) out[gr * 10 + ln] = acc[v] + bias;
            }
        }
    }
}

extern "C" void kernel_launch(void* const* d_in, const int* in_sizes, int n_in,
                              void* d_out, int out_size, void* d_ws, size_t ws_size,
                              hipStream_t stream) {
    const float* x      = (const float*)d_in[0];
    const float* conv_w = (const float*)d_in[1];
    const float* w1     = (const float*)d_in[2];
    const float* b1     = (const float*)d_in[3];
    const float* w2     = (const float*)d_in[4];
    const float* b2     = (const float*)d_in[5];
    float* out = (float*)d_out;

    unsigned short* w1T = (unsigned short*)d_ws;                       // 304*704*2 = 428032 B
    unsigned short* w2T = (unsigned short*)((char*)d_ws + N1P * K1P * 2); // 16*320*2 = 10240 B

    const int B = in_sizes[0] / 784;

    prep_w1_kernel<<<(N1P * K1P + 255) / 256, 256, 0, stream>>>(w1, w1T);
    prep_w2_kernel<<<(16 * K2P + 255) / 256, 256, 0, stream>>>(w2, w2T);

    const int nBlocks = (B + TILE_M - 1) / TILE_M;
    const size_t ldsBytes = (size_t)(TILE_M * K1P + TILE_M * K2P) * sizeof(unsigned short); // 256 KB
    fused_digit_model_kernel<<<nBlocks, 256, ldsBytes, stream>>>(
        x, conv_w, b1, b2, w1T, w2T, out, B);
}
